// dy_mixprop_1185410973782
// MI455X (gfx1250) — matrix operand order repaired and checked by reading, not compiled, not run on hardware
//
#include <hip/hip_runtime.h>

// ---------------------------------------------------------------------------
// dy_mixprop fused kernel for MI455X (gfx1250, wave32, WMMA bf16 16x16x32)
// One workgroup (256 threads = 8 waves) per (n,l) pair; 16*96 = 1536 blocks.
// Entire exp(adj) [256x256] lives in LDS (bf16, padded) -> zero global traffic
// for the intermediate graph; HBM traffic ~= read x + write out (~100 MB).
// ---------------------------------------------------------------------------

typedef __attribute__((ext_vector_type(16))) __bf16 v16bf;
typedef __attribute__((ext_vector_type(8)))  __bf16 v8bf;
typedef __attribute__((ext_vector_type(8)))  float  v8f;

#define C_IN   32
#define VDIM   256
#define LDIM   96
#define NBATCH 16
#define LDE    264            // padded bf16 row stride for E (bank stagger)
#define ALPHA_ 0.05f

// LDS byte offsets (total 268800 B < 320 KB/WGP)
#define OFF_E     0           // 256*264 bf16                = 135168
#define OFF_XS    135168      // x slab   f32 [32][256]      =  32768
#define OFF_R1    167936      // region1: X1/HA + X2/HB f32  =  65536
#define OFF_RINV  233472      // f32[256]
#define OFF_DINV  234496      // f32[256]
#define OFF_WL1   235520      // f32[1024]
#define OFF_WL2   239616      // f32[1024]
#define OFF_WM1   243712      // f32[3072]
#define OFF_WM2   256000      // f32[3072]
#define OFF_BL1   268288      // f32[32] x4 bias blocks
#define OFF_BL2   268416
#define OFF_BM1   268544
#define OFF_BM2   268672
#define SMEM_BYTES 268800

__device__ __forceinline__ v8f wmma_bf16(v16bf a, v16bf b, v8f c) {
    // D = A(16x32 bf16) * B(32x16 bf16) + C(16x16 f32)
    return __builtin_amdgcn_wmma_f32_16x16x32_bf16(
        /*neg_a=*/false, a, /*neg_b=*/false, b,
        /*c_mod=*/(short)0, c, /*reuse_a=*/false, /*reuse_b=*/false);
}

// Branchless tanh: v_exp_f32 + v_rcp_f32 + v_bfi, no divergent clauses and
// no IEEE v_div_scale/v_div_fixup ladder. e^{-2|x|} in (0,1] -> no overflow.
__device__ __forceinline__ float fast_tanh(float v) {
    float ax = fabsf(v);
    float t  = __expf(-2.0f * ax);
    float r  = (1.0f - t) * __builtin_amdgcn_rcpf(1.0f + t);
    return copysignf(r, v);
}

// A fragment, row-major f32 source: A[m][k] = S[(m0+mrow)*ld + k0 + k]
// k order used here (contiguous, same as B): element e -> k = e + 16*half
__device__ __forceinline__ v16bf ldA_rm_f32(const float* S, int ld, int m0,
                                            int k0, int mrow, int hf) {
    v16bf a;
    const float* row = S + (m0 + mrow) * ld + k0;
#pragma unroll
    for (int e = 0; e < 16; ++e) {
        int k = e + (hf << 4);
        a[e] = (__bf16)row[k];
    }
    return a;
}

// A fragment with per-k scale (folds the softmax denominator into h)
__device__ __forceinline__ v16bf ldA_rm_f32_scaled(const float* S,
                                                   const float* scale, int ld,
                                                   int m0, int k0, int mrow,
                                                   int hf) {
    v16bf a;
    const float* row = S + (m0 + mrow) * ld;
#pragma unroll
    for (int e = 0; e < 16; ++e) {
        int k = k0 + e + (hf << 4);
        a[e] = (__bf16)(row[k] * scale[k]);
    }
    return a;
}

// A fragment, column-major f32 source (for adj = x1^T * x2):
// A[m][k] = S[k*ld + (m0+mrow)]
__device__ __forceinline__ v16bf ldA_cm_f32(const float* S, int ld, int m0,
                                            int k0, int mrow, int hf) {
    v16bf a;
#pragma unroll
    for (int e = 0; e < 16; ++e) {
        int k = k0 + e + (hf << 4);
        a[e] = (__bf16)S[k * ld + m0 + mrow];
    }
    return a;
}

// B fragment, row-major f32: B[k][n] = S[k*ld + n0 + ncol]; e -> k = e + 16*half
__device__ __forceinline__ v16bf ldB_rm_f32(const float* S, int ld, int k0,
                                            int n0, int ncol, int hf) {
    v16bf b;
#pragma unroll
    for (int e = 0; e < 16; ++e) {
        int k = k0 + e + (hf << 4);
        b[e] = (__bf16)S[k * ld + n0 + ncol];
    }
    return b;
}

// B fragment, row-major bf16 (E path)
__device__ __forceinline__ v16bf ldB_rm_bf16(const __bf16* S, int ld, int k0,
                                             int n0, int ncol, int hf) {
    v16bf b;
#pragma unroll
    for (int e = 0; e < 16; ++e) {
        int k = k0 + e + (hf << 4);
        b[e] = S[k * ld + n0 + ncol];
    }
    return b;
}

// B fragment, column-major bf16 (E^T path): contiguous 2x16B -> ds_load_b128 x2
__device__ __forceinline__ v16bf ldB_cm_bf16(const __bf16* S, int ld, int k0,
                                             int n0, int ncol, int hf) {
    const __bf16* p = S + (n0 + ncol) * ld + k0 + (hf << 4);
    v8bf lo = *(const v8bf*)p;
    v8bf hi = *(const v8bf*)(p + 8);
    v16bf b;
#pragma unroll
    for (int e = 0; e < 8; ++e) { b[e] = lo[e]; b[e + 8] = hi[e]; }
    return b;
}

__launch_bounds__(256, 1)
__global__ void dy_mixprop_fused(const float* __restrict__ x,
                                 const float* __restrict__ lw1,
                                 const float* __restrict__ lb1,
                                 const float* __restrict__ lw2,
                                 const float* __restrict__ lb2,
                                 const float* __restrict__ mw1,
                                 const float* __restrict__ mb1,
                                 const float* __restrict__ mw2,
                                 const float* __restrict__ mb2,
                                 float* __restrict__ out) {
    extern __shared__ __align__(16) char smem[];
    __bf16* E    = (__bf16*)(smem + OFF_E);     // exp(adj) [256][264]
    float*  Xs   = (float*)(smem + OFF_XS);     // x slab [32][256]
    float*  HA   = (float*)(smem + OFF_R1);     // X1 / h ping  [32][256]
    float*  HB   = HA + 32 * 256;               // X2 / h pong  [32][256]
    float*  Rinv = (float*)(smem + OFF_RINV);   // 1/rowsum(E)
    float*  Dinv = (float*)(smem + OFF_DINV);   // 1/colsum(E)
    float*  Wl1  = (float*)(smem + OFF_WL1);
    float*  Wl2  = (float*)(smem + OFF_WL2);
    float*  Wm1  = (float*)(smem + OFF_WM1);
    float*  Wm2  = (float*)(smem + OFF_WM2);
    float*  Bl1  = (float*)(smem + OFF_BL1);
    float*  Bl2  = (float*)(smem + OFF_BL2);
    float*  Bm1  = (float*)(smem + OFF_BM1);
    float*  Bm2  = (float*)(smem + OFF_BM2);

    const int t    = threadIdx.x;
    const int wv   = t >> 5;          // wave 0..7
    const int lane = t & 31;
    const int hf   = lane >> 4;       // lane half
    const int lm   = lane & 15;       // mrow / ncol within tile

    const int nl   = blockIdx.x;      // 0 .. N*L-1
    const int nIdx = nl / LDIM;
    const int lIdx = nl - nIdx * LDIM;
    const float* xb = x + (size_t)nIdx * C_IN * VDIM * LDIM + lIdx;

    // ---- P0: stage weights + x slab into LDS -----------------------------
    // Warm L2 for the strided x gather (lowers to global_prefetch_b8).
    __builtin_prefetch(xb + (size_t)t * LDIM, 0, 0);
    for (int i = t; i < 1024; i += 256) { Wl1[i] = lw1[i]; Wl2[i] = lw2[i]; }
    for (int i = t; i < 3072; i += 256) { Wm1[i] = mw1[i]; Wm2[i] = mw2[i]; }
    if (t < 32) { Bl1[t] = lb1[t]; Bl2[t] = lb2[t]; Bm1[t] = mb1[t]; Bm2[t] = mb2[t]; }
    for (int i = t; i < C_IN * VDIM; i += 256) Xs[i] = xb[(size_t)i * LDIM];
    __syncthreads();

    // ---- P1: x1 = tanh(W1 x + b1), x2 = tanh(W2 x + b2)  (64 WMMA tiles) -
    for (int i = 0; i < 8; ++i) {
        int tile = wv * 8 + i;                 // 0..63
        const float* W  = (tile < 32) ? Wl1 : Wl2;
        const float* Bb = (tile < 32) ? Bl1 : Bl2;
        float*       X  = (tile < 32) ? HA : HB;
        int rm = tile & 31, mt = rm >> 4, nt = rm & 15;
        v16bf a = ldA_rm_f32(W, 32, mt * 16, 0, lm, hf);
        v16bf b = ldB_rm_f32(Xs, VDIM, 0, nt * 16, lm, hf);
        v8f acc = {};
        acc = wmma_bf16(a, b, acc);
#pragma unroll
        for (int r = 0; r < 8; ++r) {
            int m = mt * 16 + r + hf * 8;
            X[m * VDIM + nt * 16 + lm] = fast_tanh(acc[r] + Bb[m]);
        }
    }
    __syncthreads();

    // ---- P2: E = exp(x1^T x2)  (256 WMMA tiles, 16x16 grid) --------------
    // Plain exp is exact for softmax (max-subtraction cancels); adj is tiny.
    for (int i = 0; i < 32; ++i) {
        int tile = wv * 32 + i;
        int vt = tile >> 4, wt = tile & 15;
        v16bf a = ldA_cm_f32(HA, VDIM, vt * 16, 0, lm, hf);   // x1 transposed
        v16bf b = ldB_rm_f32(HB, VDIM, 0, wt * 16, lm, hf);   // x2
        v8f acc = {};
        acc = wmma_bf16(a, b, acc);
#pragma unroll
        for (int r = 0; r < 8; ++r) {
            int vv = vt * 16 + r + hf * 8;
            E[vv * LDE + wt * 16 + lm] = (__bf16)__expf(acc[r]);
        }
    }
    __syncthreads();

    // ---- P3: softmax denominators: Rinv = 1/rowsum, Dinv = 1/colsum ------
    {
        float rs = 0.f, cs = 0.f;
        const v8bf* Er = (const v8bf*)(E + t * LDE);  // 528B stride, 16B align
        for (int w8 = 0; w8 < VDIM / 8; ++w8) {
            v8bf c = Er[w8];
#pragma unroll
            for (int e = 0; e < 8; ++e) rs += (float)c[e];
        }
        for (int v = 0; v < VDIM; ++v) cs += (float)E[v * LDE + t];
        Rinv[t] = __builtin_amdgcn_rcpf(rs);
        Dinv[t] = __builtin_amdgcn_rcpf(cs);
    }
    __syncthreads();

    // ---- P4: two propagation paths + fused output MLP --------------------
    // out = sum_p sum_k Wm_p[:,32k:32k+32] @ h_k^(p)   (accumulated in regs)
    v8f accOut[4];
#pragma unroll
    for (int i = 0; i < 4; ++i) accOut[i] = (v8f){};

    for (int p = 0; p < 2; ++p) {
        const float* Wm = p ? Wm2 : Wm1;
        const float* sc = p ? Dinv : Rinv;   // fold softmax denom into h

        // phase A: mlp block 0 (h0 = x) + step1: HA = a*x + (1-a)*(x.sc)@E(^T)
        for (int i = 0; i < 4; ++i) {
            int tile = wv * 4 + i, mt = tile >> 4, nt = tile & 15;
            v16bf a = ldA_rm_f32(Wm, 96, mt * 16, 0, lm, hf);
            v16bf b = ldB_rm_f32(Xs, VDIM, 0, nt * 16, lm, hf);
            accOut[i] = wmma_bf16(a, b, accOut[i]);
        }
        for (int i = 0; i < 4; ++i) {
            int tile = wv * 4 + i, mt = tile >> 4, nt = tile & 15;
            v8f acc = {};
#pragma unroll
            for (int k0 = 0; k0 < VDIM; k0 += 32) {
                v16bf a = ldA_rm_f32_scaled(Xs, sc, VDIM, mt * 16, k0, lm, hf);
                v16bf b = p ? ldB_cm_bf16(E, LDE, k0, nt * 16, lm, hf)
                            : ldB_rm_bf16(E, LDE, k0, nt * 16, lm, hf);
                acc = wmma_bf16(a, b, acc);
            }
#pragma unroll
            for (int r = 0; r < 8; ++r) {
                int m = mt * 16 + r + hf * 8, wc = nt * 16 + lm;
                HA[m * VDIM + wc] =
                    ALPHA_ * Xs[m * VDIM + wc] + (1.0f - ALPHA_) * acc[r];
            }
        }
        __syncthreads();

        // phase B: step2: HB = a*x + (1-a)*(HA.sc)@E(^T); mlp block 1 (h1=HA)
        for (int i = 0; i < 4; ++i) {
            int tile = wv * 4 + i, mt = tile >> 4, nt = tile & 15;
            v8f acc = {};
#pragma unroll
            for (int k0 = 0; k0 < VDIM; k0 += 32) {
                v16bf a = ldA_rm_f32_scaled(HA, sc, VDIM, mt * 16, k0, lm, hf);
                v16bf b = p ? ldB_cm_bf16(E, LDE, k0, nt * 16, lm, hf)
                            : ldB_rm_bf16(E, LDE, k0, nt * 16, lm, hf);
                acc = wmma_bf16(a, b, acc);
            }
#pragma unroll
            for (int r = 0; r < 8; ++r) {
                int m = mt * 16 + r + hf * 8, wc = nt * 16 + lm;
                HB[m * VDIM + wc] =
                    ALPHA_ * Xs[m * VDIM + wc] + (1.0f - ALPHA_) * acc[r];
            }
        }
        for (int i = 0; i < 4; ++i) {
            int tile = wv * 4 + i, mt = tile >> 4, nt = tile & 15;
            v16bf a = ldA_rm_f32(Wm, 96, mt * 16, 32, lm, hf);
            v16bf b = ldB_rm_f32(HA, VDIM, 0, nt * 16, lm, hf);
            accOut[i] = wmma_bf16(a, b, accOut[i]);
        }
        __syncthreads();

        // phase C: mlp block 2 (h2 = HB)
        for (int i = 0; i < 4; ++i) {
            int tile = wv * 4 + i, mt = tile >> 4, nt = tile & 15;
            v16bf a = ldA_rm_f32(Wm, 96, mt * 16, 64, lm, hf);
            v16bf b = ldB_rm_f32(HB, VDIM, 0, nt * 16, lm, hf);
            accOut[i] = wmma_bf16(a, b, accOut[i]);
        }
        __syncthreads();
    }

    // ---- store: out[n, o, w, l] = acc + mb1[o] + mb2[o] -------------------
    float* outb = out + (size_t)nIdx * C_IN * VDIM * LDIM + lIdx;
#pragma unroll
    for (int i = 0; i < 4; ++i) {
        int tile = wv * 4 + i, mt = tile >> 4, nt = tile & 15;
#pragma unroll
        for (int r = 0; r < 8; ++r) {
            int o = mt * 16 + r + hf * 8, wc = nt * 16 + lm;
            outb[(size_t)(o * VDIM + wc) * LDIM] = accOut[i][r] + Bm1[o] + Bm2[o];
        }
    }
}

extern "C" void kernel_launch(void* const* d_in, const int* in_sizes, int n_in,
                              void* d_out, int out_size, void* d_ws,
                              size_t ws_size, hipStream_t stream) {
    (void)in_sizes; (void)n_in; (void)out_size; (void)d_ws; (void)ws_size;
    const float* x   = (const float*)d_in[0];
    const float* lw1 = (const float*)d_in[1];
    const float* lb1 = (const float*)d_in[2];
    const float* lw2 = (const float*)d_in[3];
    const float* lb2 = (const float*)d_in[4];
    const float* mw1 = (const float*)d_in[5];
    const float* mb1 = (const float*)d_in[6];
    const float* mw2 = (const float*)d_in[7];
    const float* mb2 = (const float*)d_in[8];
    float* outp = (float*)d_out;

    (void)hipFuncSetAttribute((const void*)dy_mixprop_fused,
                              hipFuncAttributeMaxDynamicSharedMemorySize,
                              SMEM_BYTES);
    dy_mixprop_fused<<<NBATCH * LDIM, 256, SMEM_BYTES, stream>>>(
        x, lw1, lb1, lw2, lb2, mw1, mb1, mw2, mb2, outp);
}
